// QuantumHybridKernelClassifier_65481071406559
// MI455X (gfx1250) — matrix operand order repaired and checked by reading, not compiled, not run on hardware
//
#include <hip/hip_runtime.h>
#include <hip/hip_bf16.h>

// ---------- types ----------
typedef float  v8f  __attribute__((ext_vector_type(8)));
typedef float  v2f  __attribute__((ext_vector_type(2)));
typedef __bf16 v16bf __attribute__((ext_vector_type(16)));
typedef unsigned int v4u __attribute__((ext_vector_type(4)));
typedef int v8i __attribute__((ext_vector_type(8)));
typedef int v4i __attribute__((ext_vector_type(4)));

#define N_DIM 4096
#define D_DIM 64

// GEMM tiling
#define KTILE 64                  // K elements staged per pipeline stage
#define ROW_BYTES 144             // 128B of bf16 data + 16B TDM pad -> 36 DWORDs/row
                                  // gcd(36,64)=4 => 16 half-wave lanes hit distinct banks

// ---------- TDM: 2D tile load, global -> LDS (bf16 elements, data_size=1 -> 2B) ----
// Descriptor packing per CDNA5 ISA (D# groups 0/1).
// flags_w0: extra bits OR'd into group1 word0 (pad enable/interval/amount).
__device__ __forceinline__ void tdm_load_tile_2d(unsigned lds_off, const void* gaddr,
                                                 unsigned tile_x, unsigned tile_y,
                                                 unsigned tensor_x, unsigned tensor_y,
                                                 unsigned stride0_elems,
                                                 unsigned flags_w0) {
  unsigned long long ga = (unsigned long long)gaddr;
  v4u g0;
  g0[0] = 1u;                                    // count=1, user mode, gather off
  g0[1] = lds_off;                               // lds_addr (bytes)
  g0[2] = (unsigned)ga;                          // global_addr[31:0]
  g0[3] = (unsigned)(ga >> 32) | (2u << 30);     // global_addr[56:32] | type=2
  v8i g1;
  g1[0] = (int)((1u << 16) | flags_w0);          // wg_mask=0, data_size=1 (2B), pads
  g1[1] = (int)((tensor_x & 0xFFFFu) << 16);     // barrier_addr=0 | tensor_dim0[15:0]
  g1[2] = (int)((tensor_x >> 16) | ((tensor_y & 0xFFFFu) << 16)); // dim0[31:16]|dim1[15:0]
  g1[3] = (int)((tensor_y >> 16) | (tile_x << 16));               // dim1[31:16]|tile_dim0
  g1[4] = (int)tile_y;                           // tile_dim1 | tile_dim2=0
  g1[5] = (int)stride0_elems;                    // tensor_dim0_stride[31:0]
  g1[6] = 0;                                     // stride0[47:32] | stride1[15:0]
  g1[7] = 0;                                     // stride1[47:16]
  v4i z4 = {0, 0, 0, 0};
#if defined(__clang_major__) && __clang_major__ >= 23
  v8i z8 = {0, 0, 0, 0, 0, 0, 0, 0};
  __builtin_amdgcn_tensor_load_to_lds(g0, g1, z4, z4, z8, 0);
#else
  __builtin_amdgcn_tensor_load_to_lds(g0, g1, z4, z4, 0);
#endif
}

// pad_enable=1 (bit20), pad_interval=4 -> 32 DWORDs (bits 24:22), pad_amount=3 -> 4 DWORDs
#define TDM_PAD_FLAGS ((1u << 20) | (4u << 22) | (3u << 25))

__device__ __forceinline__ unsigned lds_offset_of(const void* p) {
  // Low 32 bits of a flat pointer into the LDS aperture are the LDS byte offset.
  return (unsigned)(unsigned long long)p;
}

// ---------- kernel 1: row squared norms ----------
__global__ __launch_bounds__(256) void row_norms_kernel(const float* __restrict__ X,
                                                        float* __restrict__ sx) {
  int i = blockIdx.x * 256 + threadIdx.x;
  if (i < N_DIM) {
    const float4* p = (const float4*)(X + (size_t)i * D_DIM);
    float s = 0.f;
#pragma unroll
    for (int j = 0; j < D_DIM / 4; ++j) {
      float4 v = p[j];
      s += v.x * v.x + v.y * v.y + v.z * v.z + v.w * v.w;
    }
    sx[i] = s;
  }
}

// ---------- kernel 2: f32 -> bf16 convert (4 elems/thread) ----------
__global__ __launch_bounds__(256) void f32_to_bf16_kernel(const float* __restrict__ in,
                                                          __bf16* __restrict__ out,
                                                          size_t count) {
  size_t i = ((size_t)blockIdx.x * 256 + threadIdx.x) * 4;
  if (i + 3 < count) {
    float4 v = *(const float4*)(in + i);
    union { __bf16 h[4]; ushort4 s; } u;
    u.h[0] = (__bf16)v.x; u.h[1] = (__bf16)v.y;
    u.h[2] = (__bf16)v.z; u.h[3] = (__bf16)v.w;
    *(ushort4*)(out + i) = u.s;
  }
}

// ---------- kernel 3: RBF Gram via f32 WMMA, output bf16 ----------
// One wave computes one 16x16 tile of K = exp(-max(sx_i + sx_j - 2*X.X^T, 0)).
__global__ __launch_bounds__(256) void rbf_gram_kernel(const float* __restrict__ X,
                                                       const float* __restrict__ sx,
                                                       __bf16* __restrict__ Kout) {
  const int wid  = threadIdx.x >> 5;
  const int lane = threadIdx.x & 31;
  const int half = lane >> 4;
  const int l16  = lane & 15;
  const int t = blockIdx.x * 8 + wid;          // tile index over 256x256 tiles
  const int tm = t >> 8;
  const int tn = t & 255;

  const int rowA = tm * 16 + l16;              // A-matrix: lane holds row rowA
  const int rowB = tn * 16 + l16;              // B-matrix: lane holds column rowB
  const float2* Xp = (const float2*)X;         // 32 float2 per row of X

  v8f acc = {0.f, 0.f, 0.f, 0.f, 0.f, 0.f, 0.f, 0.f};
#pragma unroll
  for (int k = 0; k < D_DIM; k += 4) {
    // 32-bit A 16x4 layout: lanes 0-15 hold K=0,1 ; lanes 16-31 hold K=2,3 (per VGPR pair)
    float2 fa = Xp[(size_t)rowA * (D_DIM / 2) + (k >> 1) + half];
    float2 fb = Xp[(size_t)rowB * (D_DIM / 2) + (k >> 1) + half];
    v2f a = {fa.x, fa.y};
    v2f b = {fb.x, fb.y};
    acc = __builtin_amdgcn_wmma_f32_16x16x4_f32(false, a, false, b,
                                                (short)0, acc, false, false);
  }

  const int col = tn * 16 + l16;
  const float sxc = sx[col];
#pragma unroll
  for (int v = 0; v < 8; ++v) {
    int row = tm * 16 + half * 8 + v;          // C/D layout: lanes 16-31 are rows +8
    float d2 = sx[row] + sxc - 2.f * acc[v];
    d2 = d2 > 0.f ? d2 : 0.f;
    Kout[(size_t)row * N_DIM + col] = (__bf16)__expf(-d2);
  }
}

// ---------- kernel 4: C = relu(A @ B^T + bias) in bf16 via WMMA, TDM-staged LDS ----
// A: [4096 x 4096] bf16 row-major. B: [4096 x 4096] bf16 row-major (weights, out x in).
// Block tile 128x128, 8 waves in 4(M) x 2(N); wave tile 32x64 = 2x4 16x16 accum tiles.
// K staged in 64-wide chunks, TDM double-buffered with bank-conflict padding.
__global__ __launch_bounds__(256) void gemm_bias_relu_bf16_kernel(
    const __bf16* __restrict__ A, const __bf16* __restrict__ B,
    const float* __restrict__ bias, __bf16* __restrict__ C) {
  __shared__ alignas(16) char As[2][128 * ROW_BYTES];   // 2 x 18 KB
  __shared__ alignas(16) char Bs[2][128 * ROW_BYTES];   // 2 x 18 KB

  const int tid   = threadIdx.x;
  const int wid   = tid >> 5;
  const int lane  = tid & 31;
  const int half  = lane >> 4;
  const int l16   = lane & 15;
  const int waveM = wid & 3;                   // 0..3 -> 32 rows each
  const int waveN = wid >> 2;                  // 0..1 -> 64 cols each
  const int bm = blockIdx.y, bn = blockIdx.x;
  const int K = N_DIM;
  const int NK = K / KTILE;                    // 64 pipeline stages

  v8f acc[2][4];
  const v8f vzero = {0.f, 0.f, 0.f, 0.f, 0.f, 0.f, 0.f, 0.f};
#pragma unroll
  for (int m = 0; m < 2; ++m)
#pragma unroll
    for (int n = 0; n < 4; ++n) acc[m][n] = vzero;

  const __bf16* Abase = A + (size_t)bm * 128 * K;
  const __bf16* Bbase = B + (size_t)bn * 128 * K;

  if (wid == 0) {
    tdm_load_tile_2d(lds_offset_of(&As[0][0]), Abase, KTILE, 128, K, N_DIM, K,
                     TDM_PAD_FLAGS);
    tdm_load_tile_2d(lds_offset_of(&Bs[0][0]), Bbase, KTILE, 128, K, N_DIM, K,
                     TDM_PAD_FLAGS);
  }

  for (int kt = 0; kt < NK; ++kt) {
    const int cur = kt & 1;
    if (wid == 0) __builtin_amdgcn_s_wait_tensorcnt(0);   // buf[cur] DMA complete
    __syncthreads();                                       // visible to all waves
    if (wid == 0 && (kt + 1) < NK) {                       // prefetch next chunk
      tdm_load_tile_2d(lds_offset_of(&As[cur ^ 1][0]),
                       Abase + (size_t)(kt + 1) * KTILE, KTILE, 128, K, N_DIM, K,
                       TDM_PAD_FLAGS);
      tdm_load_tile_2d(lds_offset_of(&Bs[cur ^ 1][0]),
                       Bbase + (size_t)(kt + 1) * KTILE, KTILE, 128, K, N_DIM, K,
                       TDM_PAD_FLAGS);
    }

    union FragU { uint4 u[2]; v16bf v; };
#pragma unroll
    for (int ks = 0; ks < 2; ++ks) {           // two 32-wide K sub-chunks per stage
      v16bf afrag[2], bfrag[4];
#pragma unroll
      for (int m = 0; m < 2; ++m) {
        // 16-bit A 16x32 layout: lanes 0-15: row=l, K 0-7 & 16-23; lanes 16-31: K 8-15 & 24-31
        int row = waveM * 32 + m * 16 + l16;
        const char* rp = &As[cur][row * ROW_BYTES + ks * 64];
        FragU u;
        u.u[0] = *(const uint4*)(rp + half * 16);        // K  0- 7 /  8-15
        u.u[1] = *(const uint4*)(rp + 32 + half * 16);   // K 16-23 / 24-31
        afrag[m] = u.v;
      }
#pragma unroll
      for (int n = 0; n < 4; ++n) {
        // B 32x16: lane n holds column n; lanes 0-15 hold K 0-7 & 16-23, lanes 16-31 K 8-15 & 24-31.
        // B[k][n] = W[n][k] -> two 8-element runs of row n of the W tile, 32 bytes apart.
        int col = waveN * 64 + n * 16 + l16;
        const char* cp = &Bs[cur][col * ROW_BYTES + ks * 64 + half * 16];
        FragU u;
        u.u[0] = *(const uint4*)(cp);
        u.u[1] = *(const uint4*)(cp + 32);
        bfrag[n] = u.v;
      }
#pragma unroll
      for (int m = 0; m < 2; ++m)
#pragma unroll
        for (int n = 0; n < 4; ++n)
          acc[m][n] = __builtin_amdgcn_wmma_f32_16x16x32_bf16(
              false, afrag[m], false, bfrag[n], (short)0, acc[m][n], false, false);
    }

    __syncthreads();             // all waves done with buf[cur] before it is reloaded
  }

  // epilogue: bias + relu, store bf16
#pragma unroll
  for (int n = 0; n < 4; ++n) {
    int gcol = bn * 128 + waveN * 64 + n * 16 + l16;
    float bv = bias[gcol];
#pragma unroll
    for (int m = 0; m < 2; ++m) {
#pragma unroll
      for (int v = 0; v < 8; ++v) {
        int grow = bm * 128 + waveM * 32 + m * 16 + half * 8 + v;
        float x = acc[m][n][v] + bv;
        x = x > 0.f ? x : 0.f;
        C[(size_t)grow * N_DIM + gcol] = (__bf16)x;
      }
    }
  }
}

// ---------- kernel 5: logits head out[i,c] = sum_k h2[i,k]*Wh[c,k] + bh[c] ----------
__global__ __launch_bounds__(256) void head_kernel(const __bf16* __restrict__ H,
                                                   const float* __restrict__ Wh,
                                                   const float* __restrict__ bh,
                                                   float* __restrict__ out) {
  __shared__ float r0[256];
  __shared__ float r1[256];
  const int row = blockIdx.x;
  float s0 = 0.f, s1 = 0.f;
  for (int k = threadIdx.x; k < N_DIM; k += 256) {
    float h = (float)H[(size_t)row * N_DIM + k];
    s0 += h * Wh[k];
    s1 += h * Wh[N_DIM + k];
  }
  r0[threadIdx.x] = s0; r1[threadIdx.x] = s1;
  __syncthreads();
  for (int s = 128; s > 0; s >>= 1) {
    if (threadIdx.x < s) {
      r0[threadIdx.x] += r0[threadIdx.x + s];
      r1[threadIdx.x] += r1[threadIdx.x + s];
    }
    __syncthreads();
  }
  if (threadIdx.x == 0) {
    out[row * 2 + 0] = r0[0] + bh[0];
    out[row * 2 + 1] = r1[0] + bh[1];
  }
}

// ---------- host launch ----------
extern "C" void kernel_launch(void* const* d_in, const int* in_sizes, int n_in,
                              void* d_out, int out_size, void* d_ws, size_t ws_size,
                              hipStream_t stream) {
  const float* X  = (const float*)d_in[0];
  const float* W1 = (const float*)d_in[1];
  const float* b1 = (const float*)d_in[2];
  const float* W2 = (const float*)d_in[3];
  const float* b2 = (const float*)d_in[4];
  const float* Wh = (const float*)d_in[5];
  const float* bh = (const float*)d_in[6];
  float* out = (float*)d_out;

  const size_t n = N_DIM;
  const size_t matBF = n * n * sizeof(__bf16);        // 32 MB
  char* ws = (char*)d_ws;
  __bf16* W1b = (__bf16*)(ws);                        // [0, 32MB)
  __bf16* W2b = (__bf16*)(ws + matBF);                // [32, 64MB)
  __bf16* Kb  = (__bf16*)(ws + 2 * matBF);            // K, later reused as h2
  __bf16* H1  = (__bf16*)(ws + 3 * matBF);            // h1
  float*  sx  = (float*)(ws + 4 * matBF);             // 16 KB row norms

  row_norms_kernel<<<N_DIM / 256, 256, 0, stream>>>(X, sx);
  f32_to_bf16_kernel<<<(n * n) / 1024, 256, 0, stream>>>(W1, W1b, n * n);
  f32_to_bf16_kernel<<<(n * n) / 1024, 256, 0, stream>>>(W2, W2b, n * n);

  rbf_gram_kernel<<<(256 * 256) / 8, 256, 0, stream>>>(X, sx, Kb);

  dim3 gg(N_DIM / 128, N_DIM / 128);
  gemm_bias_relu_bf16_kernel<<<gg, 256, 0, stream>>>(Kb, W1b, b1, H1);
  gemm_bias_relu_bf16_kernel<<<gg, 256, 0, stream>>>(H1, W2b, b2, Kb /* h2 */);

  head_kernel<<<N_DIM, 256, 0, stream>>>(Kb, Wh, bh, out);
}
